// WcustomSRN_40535901340158
// MI455X (gfx1250) — matrix operand order repaired and checked by reading, not compiled, not run on hardware
//
#include <hip/hip_runtime.h>
#include <hip/hip_bf16.h>
#include <cstdint>

// ---------------------------------------------------------------------------
// Sizes (fixed by the reference)
// ---------------------------------------------------------------------------
#define BATCH   4096
#define SEQ     50
#define STEPS   49
#define IDIM    64
#define HDIM    1024
#define ODIM    128
#define EPS_F   0.1f

typedef __attribute__((ext_vector_type(16))) _Float16 v16h;
typedef __attribute__((ext_vector_type(8)))  _Float16 v8h;
typedef __attribute__((ext_vector_type(8)))  float    v8f;

// Native CDNA5 V_TANH_F32 if the builtin exists; libm fallback otherwise.
#if __has_builtin(__builtin_amdgcn_tanhf)
#define TANHF(x) __builtin_amdgcn_tanhf(x)
#elif __has_builtin(__builtin_amdgcn_tanh_f32)
#define TANHF(x) __builtin_amdgcn_tanh_f32(x)
#else
#define TANHF(x) tanhf(x)
#endif

// ---------------------------------------------------------------------------
// Workspace layout (bytes, 256-aligned)
// ---------------------------------------------------------------------------
#define OFF_SCALE 0u
#define OFF_UT    256u                                   // 1024*64*2   = 131072
#define OFF_WNT   (OFF_UT   + 131072u)                   // 1024*1024*2 = 2097152
#define OFF_FCW   (OFF_WNT  + 2097152u)                  // 128*1024*2  = 262144
#define OFF_XBF   (OFF_FCW  + 262144u)                   // 4096*50*64*2= 26214400
#define OFF_PAD   (OFF_XBF  + 26214400u)                 // 4096*4      = 16384
#define OFF_H0    (OFF_PAD  + 16384u)                    // 4096*1024*2 = 8388608
#define OFF_H1    (OFF_H0   + 8388608u)
#define OFF_HSEL  (OFF_H1   + 8388608u)                  // 4096*1024*4 = 16777216

// ---------------------------------------------------------------------------
// Fragment layouts (CDNA5 WMMA f16 16x16x32, ISA 7.12.2)
// A (16x32): lane l<16 : row l,   K = k0+{0..7} and k0+16+{0..7}
//            lane l>=16: row l-16, K = k0+8+{0..7} and k0+24+{0..7}
// B (32x16): lane l<16 : col l,   K = k0+{0..7} and k0+16+{0..7}
//            lane l>=16: col l-16, K = k0+8+{0..7} and k0+24+{0..7}
// ---------------------------------------------------------------------------
__device__ __forceinline__ v16h load_b_frag_f16(const _Float16* __restrict__ wT,
                                                int row_stride, int n0, int k0, int lane) {
  const int lh = lane & 15, hi = lane >> 4;
  const _Float16* p = wT + (size_t)(n0 + lh) * row_stride + k0 + hi * 8;
  union { v16h v; v8h h[2]; } u;
  u.h[0] = *(const v8h*)(p);
  u.h[1] = *(const v8h*)(p + 16);
  return u.v;
}

// Async global->LDS copy of 16 bytes (ASYNCcnt tracked, no VGPR round trip).
__device__ __forceinline__ void async_copy_b128(uint32_t lds_byte_off, const void* gptr) {
  asm volatile("global_load_async_to_lds_b128 %0, %1, off"
               :: "v"(lds_byte_off), "v"((unsigned long long)(uintptr_t)gptr)
               : "memory");
}
__device__ __forceinline__ void async_wait0() {
  asm volatile("s_wait_asynccnt 0" ::: "memory");
}

// ---------------------------------------------------------------------------
// Spectral norm via power iteration on W^T W (single workgroup, LDS-resident)
// ---------------------------------------------------------------------------
__global__ __launch_bounds__(1024) void power_iter_kernel(const float* __restrict__ W,
                                                          float* __restrict__ scale_out) {
  __shared__ float v[HDIM];
  __shared__ float u[HDIM];
  __shared__ float red[HDIM];
  const int t = threadIdx.x;
  v[t] = 1.0f;
  __syncthreads();

  for (int it = 0; it < 24; ++it) {
    float s = 0.0f;
    const float* row = W + (size_t)t * HDIM;
    for (int i = 0; i < HDIM; ++i) s += row[i] * v[i];
    u[t] = s;
    __syncthreads();
    float w = 0.0f;
    for (int i = 0; i < HDIM; ++i) w += W[(size_t)i * HDIM + t] * u[i];
    red[t] = w * w;
    __syncthreads();
    for (int off = 512; off > 0; off >>= 1) {
      if (t < off) red[t] += red[t + off];
      __syncthreads();
    }
    const float nrm = sqrtf(red[0]) + 1e-30f;
    v[t] = w / nrm;
    __syncthreads();
  }
  float s = 0.0f;
  const float* row = W + (size_t)t * HDIM;
  for (int i = 0; i < HDIM; ++i) s += row[i] * v[i];
  red[t] = s * s;
  __syncthreads();
  for (int off = 512; off > 0; off >>= 1) {
    if (t < off) red[t] += red[t + off];
    __syncthreads();
  }
  if (t == 0) scale_out[0] = (1.0f - EPS_F) / (sqrtf(red[0]) + 1e-30f);
}

// ---------------------------------------------------------------------------
// Prep kernels
// ---------------------------------------------------------------------------
__global__ void wnt_kernel(const float* __restrict__ W, _Float16* __restrict__ WnT,
                           const float* __restrict__ scale) {
  const int idx = blockIdx.x * 256 + threadIdx.x;          // over HDIM*HDIM
  const float s = scale[0];
  const int n = idx >> 10, k = idx & (HDIM - 1);
  WnT[idx] = (_Float16)(W[(size_t)k * HDIM + n] * s);      // WnT[n][k] = Wn[k][n]
}

__global__ void ut_kernel(const float* __restrict__ U, _Float16* __restrict__ Ut) {
  const int idx = blockIdx.x * 256 + threadIdx.x;          // over HDIM*IDIM
  const int n = idx >> 6, i = idx & (IDIM - 1);
  Ut[idx] = (_Float16)U[(size_t)i * HDIM + n];             // Ut[n][i] = U[i][n]
}

__global__ void cvt_f16_kernel(const float* __restrict__ in, _Float16* __restrict__ out, int n) {
  const int idx = blockIdx.x * 256 + threadIdx.x;
  if (idx < n) out[idx] = (_Float16)in[idx];
}

__global__ void init_h_kernel(_Float16* __restrict__ H) {
  const int idx = blockIdx.x * 256 + threadIdx.x;          // over BATCH*HDIM
  H[idx] = (_Float16)1.0f;
}

__global__ void pad_kernel(const float* __restrict__ x, int* __restrict__ padi) {
  const int b = blockIdx.x * 256 + threadIdx.x;
  if (b < BATCH) padi[b] = (int)x[((size_t)b * SEQ + (SEQ - 1)) * IDIM];
}

// ---------------------------------------------------------------------------
// One recurrence step:  Hout = tanh(X_j U + Hin Wn + b)   (fused K = 64+1024)
// Block = 256 threads = 8 waves as 2(M) x 4(N); wave tile 32x64.
// Block tile 64x256; grid (4096/64, 1024/256) = (64, 4).
// Double-buffered async pipeline: chunk c+1 is staged into the alternate LDS
// buffer via GLOBAL_LOAD_ASYNC_TO_LDS_B128 while chunk c is consumed by WMMA,
// with a single s_wait_asynccnt + barrier per chunk.
// ---------------------------------------------------------------------------
__global__ __launch_bounds__(256) void srn_step_kernel(
    const _Float16* __restrict__ Hin, _Float16* __restrict__ Hout,
    const _Float16* __restrict__ Xbf, const _Float16* __restrict__ Ut,
    const _Float16* __restrict__ WnT, const float* __restrict__ bvec,
    const int* __restrict__ padi, float* __restrict__ Hsel, int j) {
  __shared__ _Float16 sA[2][64 * 32];                      // 2 x 4 KB
  __shared__ _Float16 sB[2][256 * 32];                     // 2 x 16 KB

  const int t    = threadIdx.x;
  const int wave = t >> 5;
  const int lane = t & 31;
  const int lh   = lane & 15, hi = lane >> 4;

  const int wm = wave & 1;                                 // 0..1  (M)
  const int wn = wave >> 1;                                // 0..3  (N)
  const int blkM = blockIdx.x * 64;
  const int blkN = blockIdx.y * 256;
  const int n0w  = blkN + wn * 64;

  // Cooperative staging indices: A: thread t loads 16B (row t>>2, seg t&3);
  // B: thread t loads one 64B transposed-weight row (4 x 16B).
  const int ar = t >> 2, as = (t & 3) * 8;
  const uint32_t ldsA[2]  = { (uint32_t)(uintptr_t)(&sA[0][ar * 32 + as]),
                              (uint32_t)(uintptr_t)(&sA[1][ar * 32 + as]) };
  const uint32_t ldsB0[2] = { (uint32_t)(uintptr_t)(&sB[0][t * 32]),
                              (uint32_t)(uintptr_t)(&sB[1][t * 32]) };

  auto issue_chunk = [&](int c, int buf) {
    const _Float16* ga;
    const _Float16* gb;
    if (c < 2) {                                           // X_j @ U  (K = 64)
      const int k0 = c * 32;
      ga = Xbf + (size_t)(blkM + ar) * (SEQ * IDIM) + (size_t)j * IDIM + k0 + as;
      gb = Ut + (size_t)(blkN + t) * IDIM + k0;
    } else {                                               // Hin @ Wn (K = 1024)
      const int k0 = (c - 2) * 32;
      ga = Hin + (size_t)(blkM + ar) * HDIM + k0 + as;
      gb = WnT + (size_t)(blkN + t) * HDIM + k0;
    }
    async_copy_b128(ldsA[buf], ga);
    #pragma unroll
    for (int s = 0; s < 4; ++s) async_copy_b128(ldsB0[buf] + s * 16, gb + s * 8);
  };

  v8f acc[2][4];
  #pragma unroll
  for (int mf = 0; mf < 2; ++mf)
    #pragma unroll
    for (int nf = 0; nf < 4; ++nf) acc[mf][nf] = (v8f)0.0f;

  // Prologue: stage chunk 0.
  issue_chunk(0, 0);
  async_wait0();
  __syncthreads();

  for (int c = 0; c < 34; ++c) {
    const int buf = c & 1;
    if (c + 1 < 34) issue_chunk(c + 1, buf ^ 1);           // prefetch next chunk

    v16h af[2];
    #pragma unroll
    for (int mf = 0; mf < 2; ++mf) {
      const _Float16* pa = &sA[buf][(wm * 32 + mf * 16 + lh) * 32 + hi * 8];
      union { v16h v; v8h h[2]; } ua;
      ua.h[0] = *(const v8h*)(pa);
      ua.h[1] = *(const v8h*)(pa + 16);
      af[mf] = ua.v;
    }
    #pragma unroll
    for (int nf = 0; nf < 4; ++nf) {
      const _Float16* pb = &sB[buf][(wn * 64 + nf * 16 + lh) * 32 + hi * 8];
      union { v16h v; v8h h[2]; } ub;
      ub.h[0] = *(const v8h*)(pb);
      ub.h[1] = *(const v8h*)(pb + 16);
      #pragma unroll
      for (int mf = 0; mf < 2; ++mf) {
        acc[mf][nf] = __builtin_amdgcn_wmma_f32_16x16x32_f16(false, af[mf], false, ub.v,
                                                             (short)0, acc[mf][nf],
                                                             false, false);
      }
    }

    if (c + 1 < 34) {
      async_wait0();                                       // next chunk landed
      __syncthreads();                                     // (+ all LDS reads retired)
    }
  }

  // --- epilogue: bias, tanh (V_TANH_F32), f16 state + masked f32 capture ---
  int pr[2][8];
  #pragma unroll
  for (int mf = 0; mf < 2; ++mf)
    #pragma unroll
    for (int r = 0; r < 8; ++r) pr[mf][r] = padi[blkM + wm * 32 + mf * 16 + r + hi * 8];

  #pragma unroll
  for (int mf = 0; mf < 2; ++mf) {
    #pragma unroll
    for (int nf = 0; nf < 4; ++nf) {
      const int n = n0w + nf * 16 + lh;
      const float bias = bvec[n];
      #pragma unroll
      for (int r = 0; r < 8; ++r) {
        const int m = blkM + wm * 32 + mf * 16 + r + hi * 8;
        const float hval = TANHF(acc[mf][nf][r] + bias);
        Hout[(size_t)m * HDIM + n] = (_Float16)hval;
        if (pr[mf][r] == j) Hsel[(size_t)m * HDIM + n] = hval;
      }
    }
  }
}

// ---------------------------------------------------------------------------
// Final FC: out[b][o] = Hsel[b] . fc_W[o] + fc_b[o]   (M=4096, N=128, K=1024)
// ---------------------------------------------------------------------------
__global__ __launch_bounds__(256) void fc_final_kernel(
    const float* __restrict__ Hsel, const _Float16* __restrict__ fcWh,
    const float* __restrict__ fcb, float* __restrict__ out) {
  const int tid  = threadIdx.x;
  const int wave = tid >> 5;
  const int lane = tid & 31;
  const int lh   = lane & 15, hi = lane >> 4;

  const int m0 = blockIdx.x * 16;
  const int n0 = wave * 16;

  v8f acc = (v8f)0.0f;
  for (int k0 = 0; k0 < HDIM; k0 += 32) {
    union { v16h v; _Float16 e[16]; } a;
    const float* p0 = Hsel + (size_t)(m0 + lh) * HDIM + k0 + hi * 8;
    const float* p1 = p0 + 16;
    #pragma unroll
    for (int q = 0; q < 8; ++q) { a.e[q] = (_Float16)p0[q]; a.e[8 + q] = (_Float16)p1[q]; }
    const v16h b = load_b_frag_f16(fcWh, HDIM, n0, k0, lane);
    acc = __builtin_amdgcn_wmma_f32_16x16x32_f16(false, a.v, false, b,
                                                 (short)0, acc, false, false);
  }

  const int n = n0 + lh;
  const float bias = fcb[n];
  #pragma unroll
  for (int r = 0; r < 8; ++r) {
    const int m = m0 + r + hi * 8;
    out[(size_t)m * ODIM + n] = acc[r] + bias;
  }
}

// ---------------------------------------------------------------------------
// Host launcher
// ---------------------------------------------------------------------------
extern "C" void kernel_launch(void* const* d_in, const int* in_sizes, int n_in,
                              void* d_out, int out_size, void* d_ws, size_t ws_size,
                              hipStream_t stream) {
  const float* x    = (const float*)d_in[0];
  const float* U    = (const float*)d_in[1];
  const float* W    = (const float*)d_in[2];
  const float* bvec = (const float*)d_in[3];
  const float* fcW  = (const float*)d_in[4];
  const float* fcb  = (const float*)d_in[5];
  float* out = (float*)d_out;

  char* ws = (char*)d_ws;
  float*    scale = (float*)(ws + OFF_SCALE);
  _Float16* Ut    = (_Float16*)(ws + OFF_UT);
  _Float16* WnT   = (_Float16*)(ws + OFF_WNT);
  _Float16* fcWh  = (_Float16*)(ws + OFF_FCW);
  _Float16* Xbf   = (_Float16*)(ws + OFF_XBF);
  int*      padi  = (int*)(ws + OFF_PAD);
  _Float16* H0    = (_Float16*)(ws + OFF_H0);
  _Float16* H1    = (_Float16*)(ws + OFF_H1);
  float*    Hsel  = (float*)(ws + OFF_HSEL);

  power_iter_kernel<<<1, 1024, 0, stream>>>(W, scale);

  wnt_kernel<<<(HDIM * HDIM) / 256, 256, 0, stream>>>(W, WnT, scale);
  ut_kernel<<<(HDIM * IDIM) / 256, 256, 0, stream>>>(U, Ut);
  const int nx = BATCH * SEQ * IDIM;
  cvt_f16_kernel<<<(nx + 255) / 256, 256, 0, stream>>>(x, Xbf, nx);
  cvt_f16_kernel<<<(ODIM * HDIM + 255) / 256, 256, 0, stream>>>(fcW, fcWh, ODIM * HDIM);
  init_h_kernel<<<(BATCH * HDIM) / 256, 256, 0, stream>>>(H0);
  pad_kernel<<<(BATCH + 255) / 256, 256, 0, stream>>>(x, padi);

  _Float16* hb[2] = {H0, H1};
  dim3 grid(BATCH / 64, HDIM / 256);
  for (int j = 0; j < STEPS; ++j) {
    srn_step_kernel<<<grid, 256, 0, stream>>>(hb[j & 1], hb[(j + 1) & 1],
                                              Xbf, Ut, WnT, bvec, padi, Hsel, j);
  }

  fc_final_kernel<<<BATCH / 16, 256, 0, stream>>>(Hsel, fcWh, fcb, out);

  (void)in_sizes; (void)n_in; (void)out_size; (void)ws_size;
}
